// RowAttentionBlock_73289321939574
// MI455X (gfx1250) — compile-verified, matrix operand order checked by reading, not run on hardware
//
#include <hip/hip_runtime.h>
#include <hip/hip_bf16.h>

// ---------------- problem constants ----------------
#define B_  4
#define I_  512
#define J_  512
#define C_  512
#define P_  128
#define H_  4
#define D_  32
#define HD_ 128
#define INF_ 1.0e9f
#define EPS_ 1e-5f

typedef __attribute__((ext_vector_type(16))) _Float16 v16h;
typedef __attribute__((ext_vector_type(8)))  float    v8f;
typedef __attribute__((ext_vector_type(4)))  float    f4;
typedef __attribute__((ext_vector_type(4)))  int      i4;

union FragH { v16h v; _Float16 h[16]; unsigned u[8]; };
union FragF { v8f  v; float f[8]; };

#if __has_builtin(__builtin_amdgcn_global_load_async_to_lds_b128)
#define ASYNC_COPY 1
#else
#define ASYNC_COPY 0
#endif

// k-offset of the first element held in 32-bit reg r for a 16-bit A/B fragment
// (CDNA5 ISA 7.12.2: VGPR0 lanes0-15 K=0,1 / lanes16-31 K=8,9; VGPR4..7 = +16)
__device__ __forceinline__ int frag_kstart(int r, int half) {
    return ((r < 4) ? (2 * r) : (16 + 2 * (r - 4))) + 8 * half;
}

// load one 16x32 f16 fragment (A or B) from a row-major array.
__device__ __forceinline__ v16h load_frag(const _Float16* __restrict__ base,
                                          int row, int stride, int kbase, int half) {
    FragH f;
#pragma unroll
    for (int r = 0; r < 8; ++r) {
        int ks = kbase + frag_kstart(r, half);
        f.u[r] = *(const unsigned*)(base + (size_t)row * stride + ks);
    }
    return f.v;
}

__device__ __forceinline__ v8f wmma16(v16h a, v16h b, v8f c) {
    return __builtin_amdgcn_wmma_f32_16x16x32_f16(false, a, false, b,
                                                  (short)0, c, false, false);
}

__device__ __forceinline__ float wave_sum(float x) {
#pragma unroll
    for (int s = 16; s > 0; s >>= 1) x += __shfl_xor(x, s, 32);
    return x;
}

__device__ __forceinline__ float sigmoidf_(float x) {
    return 1.0f / (1.0f + __expf(-x));
}

// async copy of one contiguous 8 KB slab (2048 floats) into LDS; 2 b128/thread
__device__ __forceinline__ void slab_copy(const float* __restrict__ gsrc,
                                          float* lds, int tid) {
#if ASYNC_COPY
    __builtin_amdgcn_global_load_async_to_lds_b128(
        (__attribute__((address_space(1))) i4*)(gsrc + tid * 4),
        (__attribute__((address_space(3))) i4*)(lds + tid * 4), 0, 0);
    __builtin_amdgcn_global_load_async_to_lds_b128(
        (__attribute__((address_space(1))) i4*)(gsrc + 1024 + tid * 4),
        (__attribute__((address_space(3))) i4*)(lds + 1024 + tid * 4), 0, 0);
#else
    f4 a0 = __builtin_nontemporal_load((const f4*)gsrc + tid);
    f4 a1 = __builtin_nontemporal_load((const f4*)gsrc + 256 + tid);
    ((f4*)lds)[tid] = a0;
    ((f4*)lds)[256 + tid] = a1;
#endif
}

__device__ __forceinline__ void wait_async_le2(void) {
#if ASYNC_COPY
    asm volatile("s_wait_asynccnt 0x2" ::: "memory");
#endif
}
__device__ __forceinline__ void wait_async_0(void) {
#if ASYNC_COPY
    asm volatile("s_wait_asynccnt 0x0" ::: "memory");
#endif
}

// ---------------- K0a: transpose + convert weight to f16 ----------------
__global__ void wcvt_transpose(const float* __restrict__ src,
                               _Float16* __restrict__ dst, int R, int Ccols) {
    int idx = blockIdx.x * blockDim.x + threadIdx.x;
    if (idx < R * Ccols) {
        int r = idx / Ccols, c = idx % Ccols;
        dst[(size_t)c * R + r] = (_Float16)src[(size_t)r * Ccols + c];
    }
}

// ---------------- K0b: pack w_pb|w_pg into a [16][128] f16 B-panel ----------
__global__ void wp_pack(const float* __restrict__ w_pb,
                        const float* __restrict__ w_pg,
                        _Float16* __restrict__ wpT16) {
    for (int idx = threadIdx.x; idx < 16 * P_; idx += blockDim.x) {
        int n = idx >> 7, k = idx & (P_ - 1);
        float v = 0.f;
        if (n < 4)      v = w_pb[k * H_ + n];
        else if (n < 8) v = w_pg[k * H_ + (n - 4)];
        wpT16[idx] = (_Float16)v;
    }
}

// ---------------- K1: LayerNorm + two 512->128 projections (WMMA) ----------
__global__ __launch_bounds__(256) void ln_proj_kernel(
    const float* __restrict__ x, const float* __restrict__ lng,
    const float* __restrict__ lnb,
    const _Float16* __restrict__ w1T,   // [128][512]
    const _Float16* __restrict__ w2T,   // [128][512]
    float* __restrict__ ni_out,         // [B,N,C] or null
    _Float16* __restrict__ out1, float scale1,
    float* __restrict__ out2_f32,       // g or null
    _Float16* __restrict__ out2_f16,    // v or null
    const float* __restrict__ bias2) {
    __shared__ _Float16 xhat[16][C_];   // 16 KB

    const int blk = blockIdx.x;
    const int b = blk >> 5, it = blk & 31;
    const int tid = threadIdx.x, wv = tid >> 5, lane = tid & 31;
    const int half = lane >> 4, nl = lane & 15;

    float xv[2][16];
#pragma unroll
    for (int rr = 0; rr < 2; ++rr) {
        int row = 2 * wv + rr;
        const float* xr = x + ((size_t)(b * I_) + it * 16 + row) * C_;
        float s1 = 0.f, s2 = 0.f;
#pragma unroll
        for (int t = 0; t < 16; ++t) {
            float v = xr[lane + 32 * t];
            xv[rr][t] = v; s1 += v; s2 += v * v;
        }
        s1 = wave_sum(s1); s2 = wave_sum(s2);
        float mean = s1 * (1.0f / C_);
        float var  = s2 * (1.0f / C_) - mean * mean;
        float inv  = rsqrtf(var + EPS_);
#pragma unroll
        for (int t = 0; t < 16; ++t) {
            int c = lane + 32 * t;
            float xh = (xv[rr][t] - mean) * inv * lng[c] + lnb[c];
            if (ni_out) ni_out[((size_t)(b * I_) + it * 16 + row) * C_ + c] = xh;
            xhat[row][c] = (_Float16)xh;
        }
    }
    __syncthreads();

    v8f acc0 = {}, acc1 = {};
#pragma unroll 4
    for (int ks = 0; ks < 16; ++ks) {
        v16h a  = load_frag(&xhat[0][0], nl, C_, ks * 32, half);
        v16h b0 = load_frag(w1T, wv * 16 + nl, C_, ks * 32, half);
        v16h b1 = load_frag(w2T, wv * 16 + nl, C_, ks * 32, half);
        acc0 = wmma16(a, b0, acc0);
        acc1 = wmma16(a, b1, acc1);
    }
    FragF f0, f1; f0.v = acc0; f1.v = acc1;
    int col = wv * 16 + nl;
#pragma unroll
    for (int r = 0; r < 8; ++r) {
        int row = it * 16 + r + 8 * half;
        size_t off = ((size_t)(b * I_) + row) * HD_ + col;
        out1[off] = (_Float16)(f0.f[r] * scale1);
        if (out2_f32) out2_f32[off] = sigmoidf_(f1.f[r] + bias2[col]);
        else          out2_f16[off] = (_Float16)f1.f[r];
    }
}

// ---------------- K2a: streaming pair LN + gated bias + mask -> pbias -------
// One block per (b,i). Double-buffered async global->LDS copy of 8 KB slabs
// (16 j-rows x 128 P) overlaps the HBM stream with LN + WMMA.
__global__ __launch_bounds__(256) void pair_bias_kernel(
    const float* __restrict__ z, const int* __restrict__ pmask,
    const float* __restrict__ lnpg, const float* __restrict__ lnpb,
    const _Float16* __restrict__ wpT16,   // [16][128]
    float* __restrict__ pbias) {          // [B,H,I,J]
    __shared__ float    zbuf[2][16][P_];  // 16 KB (double-buffered slab)
    __shared__ _Float16 zhat[8][16][P_];  // 32 KB (8 normalized groups)

    const int blk = blockIdx.x;
    const int b = blk / I_, i = blk % I_;
    const int tid = threadIdx.x, wv = tid >> 5, lane = tid & 31;
    const int half = lane >> 4, nl = lane & 15;

    const float* zbase = z + ((size_t)b * I_ + i) * J_ * P_;
    const int*   mrow  = pmask + ((size_t)b * I_ + i) * J_;

    // per-lane LN gamma/beta (channels lane*4 .. lane*4+3, fixed per lane)
    const f4 gm = *(const f4*)(lnpg + lane * 4);
    const f4 bt = *(const f4*)(lnpb + lane * 4);

    slab_copy(zbase, &zbuf[0][0][0], tid);

    for (int g = 0; g < 32; ++g) {
        const int cur = g & 1;
        if (g + 1 < 32) {
            slab_copy(zbase + (size_t)(g + 1) * 2048, &zbuf[cur ^ 1][0][0], tid);
            wait_async_le2();   // oldest slab (the one we read now) is resident
        } else {
            wait_async_0();
        }
        __syncthreads();

        // cooperative LN: wave owns rows 2wv, 2wv+1 of this 16-row group
        const int gg = g & 7;
#pragma unroll
        for (int rr = 0; rr < 2; ++rr) {
            int row = 2 * wv + rr;
            f4 zv = *(const f4*)&zbuf[cur][row][lane * 4];
            float s1 = wave_sum(zv.x + zv.y + zv.z + zv.w);
            float s2 = wave_sum(zv.x * zv.x + zv.y * zv.y +
                                zv.z * zv.z + zv.w * zv.w);
            float mean = s1 * (1.0f / P_);
            float inv  = rsqrtf(s2 * (1.0f / P_) - mean * mean + EPS_);
            union { _Float16 h[4]; unsigned long long q; } pk;
            pk.h[0] = (_Float16)((zv.x - mean) * inv * gm.x + bt.x);
            pk.h[1] = (_Float16)((zv.y - mean) * inv * gm.y + bt.y);
            pk.h[2] = (_Float16)((zv.z - mean) * inv * gm.z + bt.z);
            pk.h[3] = (_Float16)((zv.w - mean) * inv * gm.w + bt.w);
            *(unsigned long long*)&zhat[gg][row][lane * 4] = pk.q;
        }
        __syncthreads();

        if (gg == 7) {
            // each wave: [16 pairs x 128] @ [128 x 8] for its group
            const int sg = (g - 7) + wv;
            const int j0 = sg * 16;
            v8f acc = {};
#pragma unroll
            for (int kq = 0; kq < 4; ++kq) {
                v16h a  = load_frag(&zhat[wv][0][0], nl, P_, kq * 32, half);
                v16h bw = load_frag(wpT16, nl, P_, kq * 32, half);
                acc = wmma16(a, bw, acc);
            }
            FragF fa; fa.v = acc;
#pragma unroll
            for (int r = 0; r < 8; ++r) {
                // lane n<4 holds left[h=n]; lane n+4 holds gate logit[h=n]
                float gatel = __shfl(fa.f[r], lane + 4, 32);
                if (nl < 4) {
                    int m = r + 8 * half;
                    int j = j0 + m;
                    float mb = INF_ * ((float)mrow[j] - 1.0f);
                    pbias[(((size_t)(b * H_ + nl) * I_) + i) * J_ + j] =
                        fa.f[r] * sigmoidf_(gatel) + mb;
                }
            }
            __syncthreads();   // zhat reused by next super-group
        }
    }
}

// ---------------- K2b: fused attention per (b,h,i-tile) --------------------
__global__ __launch_bounds__(256) void attn_kernel(
    const _Float16* __restrict__ q16, const _Float16* __restrict__ k16,
    const _Float16* __restrict__ v16, const float* __restrict__ pbias,
    const float* __restrict__ g32, _Float16* __restrict__ og16) {
    __shared__ float    logits[16][J_];   // 32 KB
    __shared__ float    red[16][16];      // 1 KB
    __shared__ _Float16 atile[16][J_];    // 16 KB
    __shared__ float    opart[4][16][32]; // 8 KB

    const int blk = blockIdx.x;
    const int it = blk & 31, h = (blk >> 5) & 3, b = blk >> 7;
    const int tid = threadIdx.x, wv = tid >> 5, lane = tid & 31;
    const int half = lane >> 4, nl = lane & 15;

    v16h aq = load_frag(q16 + (size_t)b * I_ * HD_, it * 16 + nl, HD_, h * D_, half);
#pragma unroll
    for (int jt = wv * 4; jt < wv * 4 + 4; ++jt) {
        FragF c;
#pragma unroll
        for (int r = 0; r < 8; ++r) {
            int m = r + 8 * half;
            c.f[r] = pbias[(((size_t)(b * H_ + h) * I_) + it * 16 + m) * J_ +
                           jt * 16 + nl];
        }
        v16h bk = load_frag(k16 + (size_t)b * J_ * HD_, jt * 16 + nl, HD_,
                            h * D_, half);
        FragF d; d.v = wmma16(aq, bk, c.v);
#pragma unroll
        for (int r = 0; r < 8; ++r)
            logits[r + 8 * half][jt * 16 + nl] = d.f[r];
    }
    __syncthreads();

    {
        int r = tid >> 4, p = tid & 15;
        float lmax = -3.0e38f;
        for (int cc = p * 32; cc < p * 32 + 32; ++cc)
            lmax = fmaxf(lmax, logits[r][cc]);
        red[r][p] = lmax; __syncthreads();
        if (p == 0) {
            float m = red[r][0];
#pragma unroll
            for (int q = 1; q < 16; ++q) m = fmaxf(m, red[r][q]);
            red[r][0] = m;
        }
        __syncthreads();
        float rmax = red[r][0];
        float lsum = 0.f;
        for (int cc = p * 32; cc < p * 32 + 32; ++cc) {
            float e = __expf(logits[r][cc] - rmax);
            logits[r][cc] = e; lsum += e;
        }
        __syncthreads();
        red[r][p] = lsum; __syncthreads();
        if (p == 0) {
            float s = 0.f;
#pragma unroll
            for (int q = 0; q < 16; ++q) s += red[r][q];
            red[r][0] = s;
        }
        __syncthreads();
        float inv = 1.0f / red[r][0];
        for (int cc = p * 32; cc < p * 32 + 32; ++cc)
            atile[r][cc] = (_Float16)(logits[r][cc] * inv);
    }
    __syncthreads();

    {
        int ntile = wv & 1, kq = wv >> 1;
        v8f acc = {};
#pragma unroll
        for (int ks = 0; ks < 4; ++ks) {
            int k0 = (kq * 4 + ks) * 32;
            v16h af = load_frag(&atile[0][0], nl, J_, k0, half);
            FragH bf;
            int dcol = ntile * 16 + nl;
#pragma unroll
            for (int e = 0; e < 16; ++e) {
                int kk = k0 + frag_kstart(e >> 1, half) + (e & 1);
                bf.h[e] = v16[(size_t)b * J_ * HD_ + (size_t)kk * HD_ +
                              h * D_ + dcol];
            }
            acc = wmma16(af, bf.v, acc);
        }
        FragF fa; fa.v = acc;
#pragma unroll
        for (int r = 0; r < 8; ++r)
            opart[kq][r + 8 * half][ntile * 16 + nl] = fa.f[r];
    }
    __syncthreads();
    for (int e = tid; e < 512; e += 256) {
        int m = e >> 5, d = e & 31;
        float s = opart[0][m][d] + opart[1][m][d] + opart[2][m][d] + opart[3][m][d];
        size_t off = ((size_t)b * I_ + it * 16 + m) * HD_ + h * D_ + d;
        og16[off] = (_Float16)(s * g32[off]);
    }
}

// ---------------- K3: out = ni + mask_i * ((o*g) @ wo + bo) ----------------
__global__ __launch_bounds__(256) void out_proj_kernel(
    const _Float16* __restrict__ og16, const _Float16* __restrict__ woT,
    const float* __restrict__ bo, const float* __restrict__ ni,
    const int* __restrict__ mask_i, float* __restrict__ out) {
    const int blk = blockIdx.x;
    const int b = blk >> 5, it = blk & 31;
    const int tid = threadIdx.x, wv = tid >> 5, lane = tid & 31;
    const int half = lane >> 4, nl = lane & 15;

    v16h afr[4];
#pragma unroll
    for (int ks = 0; ks < 4; ++ks)
        afr[ks] = load_frag(og16 + (size_t)b * I_ * HD_, it * 16 + nl, HD_,
                            ks * 32, half);
#pragma unroll
    for (int t = 0; t < 4; ++t) {
        int col = (wv * 4 + t) * 16 + nl;
        v8f acc = {};
#pragma unroll
        for (int ks = 0; ks < 4; ++ks) {
            v16h bw = load_frag(woT, col, HD_, ks * 32, half);
            acc = wmma16(afr[ks], bw, acc);
        }
        FragF fa; fa.v = acc;
#pragma unroll
        for (int r = 0; r < 8; ++r) {
            int row = it * 16 + r + 8 * half;
            size_t off = ((size_t)b * I_ + row) * C_ + col;
            float msk = (float)mask_i[b * I_ + row];
            out[off] = ni[off] + msk * (fa.f[r] + bo[col]);
        }
    }
}

// ---------------- launcher ----------------
extern "C" void kernel_launch(void* const* d_in, const int* in_sizes, int n_in,
                              void* d_out, int out_size, void* d_ws, size_t ws_size,
                              hipStream_t stream) {
    const float* node_i   = (const float*)d_in[0];
    const float* node_j   = (const float*)d_in[1];
    const float* pair     = (const float*)d_in[2];
    const int*   pmask    = (const int*)d_in[3];
    const int*   nmask_i  = (const int*)d_in[4];
    const float* ln_i_g   = (const float*)d_in[5];
    const float* ln_i_b   = (const float*)d_in[6];
    const float* ln_j_g   = (const float*)d_in[7];
    const float* ln_j_b   = (const float*)d_in[8];
    const float* ln_p_g   = (const float*)d_in[9];
    const float* ln_p_b   = (const float*)d_in[10];
    const float* w_pb     = (const float*)d_in[11];
    const float* w_pg     = (const float*)d_in[12];
    const float* wq       = (const float*)d_in[13];
    const float* wk       = (const float*)d_in[14];
    const float* wvw      = (const float*)d_in[15];
    const float* wg       = (const float*)d_in[16];
    const float* bg       = (const float*)d_in[17];
    const float* wo       = (const float*)d_in[18];
    const float* bo       = (const float*)d_in[19];
    float* out = (float*)d_out;

    char* p = (char*)d_ws;
    auto carve = [&](size_t bytes) {
        void* r = (void*)p;
        p += (bytes + 255) & ~(size_t)255;
        return r;
    };
    float*     ni    = (float*)carve((size_t)B_ * I_ * C_ * 4);
    _Float16*  q16   = (_Float16*)carve((size_t)B_ * I_ * HD_ * 2);
    _Float16*  k16   = (_Float16*)carve((size_t)B_ * J_ * HD_ * 2);
    _Float16*  v16   = (_Float16*)carve((size_t)B_ * J_ * HD_ * 2);
    float*     g32   = (float*)carve((size_t)B_ * I_ * HD_ * 4);
    _Float16*  og16  = (_Float16*)carve((size_t)B_ * I_ * HD_ * 2);
    float*     pbias = (float*)carve((size_t)B_ * H_ * I_ * J_ * 4);
    _Float16*  wqT   = (_Float16*)carve((size_t)C_ * HD_ * 2);
    _Float16*  wkT   = (_Float16*)carve((size_t)C_ * HD_ * 2);
    _Float16*  wvT   = (_Float16*)carve((size_t)C_ * HD_ * 2);
    _Float16*  wgT   = (_Float16*)carve((size_t)C_ * HD_ * 2);
    _Float16*  woT   = (_Float16*)carve((size_t)C_ * HD_ * 2);
    _Float16*  wpT16 = (_Float16*)carve((size_t)16 * P_ * 2);

    const int cvt_blocks = (C_ * HD_ + 255) / 256;
    wcvt_transpose<<<cvt_blocks, 256, 0, stream>>>(wq,  wqT, C_, HD_);
    wcvt_transpose<<<cvt_blocks, 256, 0, stream>>>(wk,  wkT, C_, HD_);
    wcvt_transpose<<<cvt_blocks, 256, 0, stream>>>(wvw, wvT, C_, HD_);
    wcvt_transpose<<<cvt_blocks, 256, 0, stream>>>(wg,  wgT, C_, HD_);
    wcvt_transpose<<<cvt_blocks, 256, 0, stream>>>(wo,  woT, HD_, C_);
    wp_pack<<<1, 256, 0, stream>>>(w_pb, w_pg, wpT16);

    const float qscale = 0.17677669529663687f;  // 1/sqrt(32)
    ln_proj_kernel<<<B_ * (I_ / 16), 256, 0, stream>>>(
        node_i, ln_i_g, ln_i_b, wqT, wgT, ni, q16, qscale, g32, nullptr, bg);
    ln_proj_kernel<<<B_ * (J_ / 16), 256, 0, stream>>>(
        node_j, ln_j_g, ln_j_b, wkT, wvT, nullptr, k16, 1.0f, nullptr, v16,
        nullptr);

    pair_bias_kernel<<<B_ * I_, 256, 0, stream>>>(pair, pmask, ln_p_g, ln_p_b,
                                                  wpT16, pbias);
    attn_kernel<<<B_ * H_ * (I_ / 16), 256, 0, stream>>>(q16, k16, v16, pbias,
                                                         g32, og16);
    out_proj_kernel<<<B_ * (I_ / 16), 256, 0, stream>>>(og16, woT, bo, ni,
                                                        nmask_i, out);
}
